// LshSelfAttention_64965675319649
// MI455X (gfx1250) — matrix operand order repaired and checked by reading, not compiled, not run on hardware
//
#include <hip/hip_runtime.h>
#include <hip/hip_bf16.h>
#include <math.h>

typedef __bf16 bf16_t;
typedef __attribute__((ext_vector_type(8)))  __bf16 v8bf;
typedef __attribute__((ext_vector_type(16))) __bf16 v16bf;
typedef __attribute__((ext_vector_type(8)))  float  v8f;

// ---- problem constants (from reference setup_inputs) ----
constexpr int kB    = 4;
constexpr int kT    = 4096;          // sequence length (2^12)
constexpr int kH    = 1024;          // model dim
constexpr int kNH   = 16;            // heads
constexpr int kD    = 64;            // dim per head
constexpr int kNHash= 2;
constexpr int kNBuck= 64;            // n_buckets = T / bucket_size
constexpr int kW    = 64;            // bucket (chunk) size
constexpr int kC    = kNHash * kNBuck;   // 128 chunks in sorted space
constexpr int kBh   = kB * kNH;          // 64 merged batch*heads
constexpr int kNT   = kNHash * kT;       // 8192 sorted slots per head

// ---------------------------------------------------------------------------
// WMMA helpers: bf16 16x16x32, f32 accumulate.  Fragment layouts per CDNA5 ISA
// 7.12.2 (wave32).
// ---------------------------------------------------------------------------
__device__ __forceinline__ v8f wmma_bf16(v16bf a, v16bf b, v8f c) {
  return __builtin_amdgcn_wmma_f32_16x16x32_bf16(
      /*neg_a=*/false, a, /*neg_b=*/false, b,
      /*c_mod=*/(short)0, c, /*reuse_a=*/false, /*reuse_b=*/false);
}

// A matrix 16x32 bf16, row-major source with leading dim ld (elements).
// lane<16:  elems 0..7 -> K = kbase+0..7,  elems 8..15 -> K = kbase+16..23
// lane>=16: elems 0..7 -> K = kbase+8..15, elems 8..15 -> K = kbase+24..31
__device__ __forceinline__ v16bf load_a_frag(const bf16_t* p, int ld,
                                             int row0, int kbase, int lane) {
  int lo = lane >> 4;
  int m  = lane & 15;
  const bf16_t* r = p + (size_t)(row0 + m) * ld + kbase + 8 * lo;
  v8bf a0 = *(const v8bf*)(r);
  v8bf a1 = *(const v8bf*)(r + 16);
  v16bf out;
#pragma unroll
  for (int e = 0; e < 8; ++e) { out[e] = a0[e]; out[e + 8] = a1[e]; }
  return out;
}

// B matrix 32x16 bf16 from transposed storage BT[N][K] (row-major, ld elems).
// lane holds N = lane&15, elems 0..7 -> K = kbase+8*(lane>>4)+e, elems 8..15 -> +16.
__device__ __forceinline__ v16bf load_b_frag(const bf16_t* p, int ld,
                                             int ncol0, int kbase, int lane) {
  int lo = lane >> 4;
  int n  = lane & 15;
  const bf16_t* r = p + (size_t)(ncol0 + n) * ld + kbase + 8 * lo;
  v8bf b0 = *(const v8bf*)(r);
  v8bf b1 = *(const v8bf*)(r + 16);
  v16bf out;
#pragma unroll
  for (int e = 0; e < 8; ++e) { out[e] = b0[e]; out[e + 8] = b1[e]; }
  return out;
}

// ---------------------------------------------------------------------------
// Prep kernels
// ---------------------------------------------------------------------------
__global__ void cvt_f32_to_bf16(const float* __restrict__ src,
                                bf16_t* __restrict__ dst, int n) {
  int i = blockIdx.x * blockDim.x + threadIdx.x;
  if (i < n) dst[i] = (bf16_t)src[i];
}

// dst[c][r] = (bf16) src[r][c]   (src rows x cols, f32)
__global__ void transpose_f32_to_bf16(const float* __restrict__ src,
                                      bf16_t* __restrict__ dst,
                                      int rows, int cols) {
  int i = blockIdx.x * blockDim.x + threadIdx.x;
  if (i >= rows * cols) return;
  int r = i / cols, c = i % cols;
  dst[(size_t)c * rows + r] = (bf16_t)src[(size_t)r * cols + c];
}

// ---------------------------------------------------------------------------
// Register-blocked bf16 WMMA GEMM: C[M,N] = A[M,K] (row-major bf16) x B, with
// B given as BT[N][K].  One wave computes a 32x64 block (2 M-tiles x 4
// N-tiles): 8 WMMAs per 2 A-frag + 4 B-frag loads.
// mode 0: scatter f32 into qk head layout  [(b*16+n)][t][d]
// mode 1: scatter bf16 into v  head layout
// mode 2: row-major f32 output [M][N]
// Requires M % 32 == 0, N % 64 == 0, K % 32 == 0.
// ---------------------------------------------------------------------------
__global__ __launch_bounds__(256) void wmma_gemm_bf16(
    const bf16_t* __restrict__ A, const bf16_t* __restrict__ BT,
    int M, int N, int K,
    float* __restrict__ outF, bf16_t* __restrict__ outB, int mode) {
  int lane = threadIdx.x & 31;
  int wv   = threadIdx.x >> 5;
  int gw   = blockIdx.x * 8 + wv;
  int blocksN = N >> 6;                  // 64-wide N blocks
  int total   = (M >> 5) * blocksN;      // 32-tall M blocks
  if (gw >= total) return;               // wave-uniform
  int tM = (gw / blocksN) << 5;
  int tN = (gw % blocksN) << 6;

  v8f zero = {0.f, 0.f, 0.f, 0.f, 0.f, 0.f, 0.f, 0.f};
  v8f acc[2][4];
#pragma unroll
  for (int mi = 0; mi < 2; ++mi)
#pragma unroll
    for (int nt = 0; nt < 4; ++nt) acc[mi][nt] = zero;

  for (int k = 0; k < K; k += 32) {
    v16bf a0 = load_a_frag(A, K, tM,      k, lane);
    v16bf a1 = load_a_frag(A, K, tM + 16, k, lane);
#pragma unroll
    for (int nt = 0; nt < 4; ++nt) {
      v16bf b = load_b_frag(BT, K, tN + nt * 16, k, lane);
      acc[0][nt] = wmma_bf16(a0, b, acc[0][nt]);
      acc[1][nt] = wmma_bf16(a1, b, acc[1][nt]);
    }
  }

  int lo = lane >> 4, nn = lane & 15;
#pragma unroll
  for (int mi = 0; mi < 2; ++mi) {
#pragma unroll
    for (int nt = 0; nt < 4; ++nt) {
#pragma unroll
      for (int j = 0; j < 8; ++j) {
        int row = tM + mi * 16 + j + 8 * lo;  // C layout: M = vgpr + 8*(lane/16)
        int col = tN + nt * 16 + nn;          //           N = lane%16
        float v = acc[mi][nt][j];
        if (mode == 2) {
          outF[(size_t)row * N + col] = v;
        } else {
          int b_ = row >> 12, t = row & (kT - 1);
          int n_ = col >> 6,  dd = col & (kD - 1);
          size_t idx = (((size_t)(b_ * kNH + n_)) * kT + t) * kD + dd;
          if (mode == 0) outF[idx] = v;
          else           outB[idx] = (bf16_t)v;
        }
      }
    }
  }
}

// ---------------------------------------------------------------------------
// LSH bucketing: rotated = qk . rot, argmax over [r, -r]; key packs
// (bucket*T + t) in bits [31:13] and i = hash*T + t in bits [12:0].
// rot layout: [d=64][hash=2][r=32]
// ---------------------------------------------------------------------------
__global__ __launch_bounds__(256) void lsh_buckets(
    const float* __restrict__ qk, const float* __restrict__ rot,
    unsigned int* __restrict__ keybuf) {
  int gid = blockIdx.x * blockDim.x + threadIdx.x;   // over Bh*T
  if (gid >= kBh * kT) return;
  int bh = gid >> 12;
  int t  = gid & (kT - 1);
  const float* qrow = qk + (size_t)gid * kD;
  float q[kD];
#pragma unroll
  for (int d = 0; d < kD; ++d) q[d] = qrow[d];
#pragma unroll
  for (int h = 0; h < kNHash; ++h) {
    int   best  = 0;
    float bestv = -3.4e38f;
    for (int r = 0; r < 32; ++r) {
      float s = 0.f;
#pragma unroll
      for (int d = 0; d < kD; ++d) s += q[d] * rot[(d * kNHash + h) * 32 + r];
      if ( s > bestv) { bestv =  s; best = r;      }
      if (-s > bestv) { bestv = -s; best = r + 32; }
    }
    int bucket = best + h * kNBuck;              // + hash offset (0..127)
    unsigned int i   = (unsigned)(h * kT + t);   // 13 bits
    unsigned int key = (unsigned)(bucket * kT + t);  // 19 bits
    keybuf[(size_t)bh * kNT + i] = (key << 13) | i;
  }
}

// ---------------------------------------------------------------------------
// Bitonic sort of 8192 packed keys per head; emit sticker[j] = original i.
// ---------------------------------------------------------------------------
__global__ __launch_bounds__(1024) void lsh_sort(
    const unsigned int* __restrict__ keybuf, int* __restrict__ sticker) {
  __shared__ unsigned int sk[kNT];
  int bh = blockIdx.x;
  for (int i = threadIdx.x; i < kNT; i += 1024)
    sk[i] = keybuf[(size_t)bh * kNT + i];
  __syncthreads();
  for (int k = 2; k <= kNT; k <<= 1) {
    for (int j = k >> 1; j > 0; j >>= 1) {
      for (int i = threadIdx.x; i < kNT; i += 1024) {
        int ixj = i ^ j;
        if (ixj > i) {
          bool up = ((i & k) == 0);
          unsigned a = sk[i], b = sk[ixj];
          if (up ? (a > b) : (a < b)) { sk[i] = b; sk[ixj] = a; }
        }
      }
      __syncthreads();
    }
  }
  for (int i = threadIdx.x; i < kNT; i += 1024)
    sticker[(size_t)bh * kNT + i] = (int)(sk[i] & 0x1FFFu);
}

// ---------------------------------------------------------------------------
// Chunked LSH attention.  One 128-thread (4-wave) workgroup per (head, chunk).
// dots = bq(64x64) x bk^T(64x128) via WMMA, penalties + logsumexp in the f32
// accumulators, AV = w(64x128) x bv(128x64) via WMMA.  Outputs scattered to
// (hash, t) slots, which is exactly the reference's undo_sort gather.
// ---------------------------------------------------------------------------
__global__ __launch_bounds__(128) void lsh_attention(
    const float* __restrict__ qk, const bf16_t* __restrict__ vbuf,
    const int* __restrict__ sticker, const unsigned char* __restrict__ pad,
    float* __restrict__ uo, float* __restrict__ ulse) {
  __shared__ __align__(16) bf16_t sQ [kW * kD];        // 64x64 queries
  __shared__ __align__(16) bf16_t sK [2 * kW * kD];    // 128x64 normalized keys
  __shared__ __align__(16) bf16_t sVT[kD * 2 * kW];    // v transposed: [d][key]
  __shared__ __align__(16) bf16_t sW [kW * 2 * kW];    // softmax weights 64x128
  __shared__ int   sIdx[kW];     // sorted original index i per query row
  __shared__ int   sKt [2 * kW]; // original position t per key
  __shared__ float sKp [2 * kW]; // padding penalty per key
  __shared__ float sLse[kW];

  int bh  = blockIdx.x >> 7;            // C = 128 chunks
  int c   = blockIdx.x & (kC - 1);
  int tid = threadIdx.x;
  int cprev = (c + kC - 1) & (kC - 1);
  const int* stk = sticker + (size_t)bh * kNT;
  int b = bh >> 4;

  // ---- gather keys / values (one thread per key row) ----
  {
    int kk = tid;                                       // 0..127
    int j2 = (kk < kW) ? (c * kW + kk) : (cprev * kW + (kk - kW));
    int i2 = stk[j2];
    int t2 = i2 & (kT - 1);
    const float* qrow = qk + ((size_t)bh * kT + t2) * kD;
    float ss = 0.f;
#pragma unroll
    for (int d = 0; d < kD; ++d) { float x = qrow[d]; ss += x * x; }
    float inv = 1.0f / fmaxf(sqrtf(ss), 1e-6f);
#pragma unroll
    for (int d = 0; d < kD; ++d) sK[kk * kD + d] = (bf16_t)(qrow[d] * inv);
    const bf16_t* vrow = vbuf + ((size_t)bh * kT + t2) * kD;
#pragma unroll
    for (int d = 0; d < kD; ++d) sVT[d * (2 * kW) + kk] = vrow[d];
    sKt[kk] = t2;
    sKp[kk] = pad[b * kT + t2] ? 1e9f : 0.f;
  }
  // ---- gather queries (threads 0..63) ----
  if (tid < kW) {
    int j = c * kW + tid;
    int i = stk[j];
    sIdx[tid] = i;
    int t = i & (kT - 1);
    const float* qrow = qk + ((size_t)bh * kT + t) * kD;
#pragma unroll
    for (int d = 0; d < kD; ++d) sQ[tid * kD + d] = (bf16_t)qrow[d];
  }
  __syncthreads();

  int wv = tid >> 5, lane = tid & 31;
  int lo = lane >> 4, nn = lane & 15;
  int M0 = wv * 16;                     // this wave's 16-row query strip

  // ---- dots: 8 N-tiles of 16x16, K = 64 (2 WMMA steps) ----
  v8f dacc[8];
  v8f zero = {0.f, 0.f, 0.f, 0.f, 0.f, 0.f, 0.f, 0.f};
#pragma unroll
  for (int nt = 0; nt < 8; ++nt) dacc[nt] = zero;
  for (int k = 0; k < kD; k += 32) {
    v16bf a = load_a_frag(sQ, kD, M0, k, lane);
#pragma unroll
    for (int nt = 0; nt < 8; ++nt) {
      v16bf bb = load_b_frag(sK, kD, nt * 16, k, lane);
      dacc[nt] = wmma_bf16(a, bb, dacc[nt]);
    }
  }

  // ---- penalties (in f32 accumulators) ----
  const float scale = 0.125f;           // 64^-0.5
#pragma unroll
  for (int nt = 0; nt < 8; ++nt) {
#pragma unroll
    for (int j = 0; j < 8; ++j) {
      int r  = M0 + j + 8 * lo;
      int kc = nt * 16 + nn;
      float dv = dacc[nt][j] * scale;
      if (sKt[kc] == (sIdx[r] & (kT - 1))) dv -= 1e5f;   // self-attention pen.
      dv -= sKp[kc];                                     // padding penalty
      dacc[nt][j] = dv;
    }
  }

  // ---- per-row logsumexp + weights (row lives in one 16-lane half) ----
#pragma unroll
  for (int j = 0; j < 8; ++j) {
    int r = M0 + j + 8 * lo;
    float m = -3.4e38f;
#pragma unroll
    for (int nt = 0; nt < 8; ++nt) m = fmaxf(m, dacc[nt][j]);
#pragma unroll
    for (int s = 1; s < 16; s <<= 1) m = fmaxf(m, __shfl_xor(m, s, 32));
    float se = 0.f;
#pragma unroll
    for (int nt = 0; nt < 8; ++nt) se += __expf(dacc[nt][j] - m);
#pragma unroll
    for (int s = 1; s < 16; s <<= 1) se += __shfl_xor(se, s, 32);
    float lse = m + __logf(se);
#pragma unroll
    for (int nt = 0; nt < 8; ++nt)
      sW[r * (2 * kW) + nt * 16 + nn] = (bf16_t)__expf(dacc[nt][j] - lse);
    if (nn == 0) sLse[r] = lse;
  }
  __syncthreads();

  // ---- AV: 4 N-tiles of 16x16, K = 128 (4 WMMA steps) ----
  v8f oacc[4];
#pragma unroll
  for (int nt = 0; nt < 4; ++nt) oacc[nt] = zero;
  for (int k = 0; k < 2 * kW; k += 32) {
    v16bf a = load_a_frag(sW, 2 * kW, M0, k, lane);
#pragma unroll
    for (int nt = 0; nt < 4; ++nt) {
      v16bf bb = load_b_frag(sVT, 2 * kW, nt * 16, k, lane);
      oacc[nt] = wmma_bf16(a, bb, oacc[nt]);
    }
  }

  // ---- scatter to (hash, t) slots: uo[bh][hash][t][d], ulse[bh][hash][t] ----
#pragma unroll
  for (int j = 0; j < 8; ++j) {
    int r = M0 + j + 8 * lo;
    int i = sIdx[r];
    int hs = i >> 12;
    int t  = i & (kT - 1);
    size_t base = (((size_t)bh * kNHash + hs) * kT + t) * kD;
#pragma unroll
    for (int nt = 0; nt < 4; ++nt) uo[base + nt * 16 + nn] = oacc[nt][j];
  }
  if (tid < kW) {
    int i = sIdx[tid];
    int hs = i >> 12;
    int t  = i & (kT - 1);
    ulse[((size_t)bh * kNHash + hs) * kT + t] = sLse[tid];
  }
}

// ---------------------------------------------------------------------------
// Combine hash rounds: softmax over the 2 LSE logits; emit bf16 directly in
// row-major [b*T+t][n*64+d] so the output projection is a plain WMMA GEMM.
// ---------------------------------------------------------------------------
__global__ __launch_bounds__(256) void lsh_combine(
    const float* __restrict__ uo, const float* __restrict__ ulse,
    bf16_t* __restrict__ headout) {
  int gid = blockIdx.x * blockDim.x + threadIdx.x;    // over Bh*T
  if (gid >= kBh * kT) return;
  int bh = gid >> 12;
  int t  = gid & (kT - 1);
  int b  = bh >> 4;
  int n  = bh & 15;
  size_t base = (size_t)bh * kNHash * kT;
  float l0 = ulse[base + t];
  float l1 = ulse[base + kT + t];
  float m  = fmaxf(l0, l1);
  float p0 = __expf(l0 - m), p1 = __expf(l1 - m);
  float inv = 1.0f / (p0 + p1);
  const float* o0 = uo + (base + t) * kD;
  const float* o1 = uo + (base + kT + t) * kD;
  bf16_t* dst = headout + ((size_t)(b * kT + t) * kH + n * kD);
#pragma unroll
  for (int d = 0; d < kD; ++d)
    dst[d] = (bf16_t)((o0[d] * p0 + o1[d] * p1) * inv);
}

// ---------------------------------------------------------------------------
// Host-side launch
// ---------------------------------------------------------------------------
extern "C" void kernel_launch(void* const* d_in, const int* in_sizes, int n_in,
                              void* d_out, int out_size, void* d_ws, size_t ws_size,
                              hipStream_t stream) {
  (void)in_sizes; (void)n_in; (void)out_size; (void)ws_size;
  const float*         q_in = (const float*)d_in[0];           // [B,T,H]
  const unsigned char* pad  = (const unsigned char*)d_in[1];   // [B,T] bool
  const float*         wqk  = (const float*)d_in[2];           // [H,NH,DPH]
  const float*         wv   = (const float*)d_in[3];           // [H,NH,DPH]
  const float*         wo   = (const float*)d_in[4];           // [NH,DPH,H]
  const float*         rot  = (const float*)d_in[5];           // [DPH,2,32]
  float* out = (float*)d_out;                                  // [B,T,H]

  // ---- workspace carve-up ----
  char* ws = (char*)d_ws;
  size_t off = 0;
  auto carve = [&](size_t bytes) {
    char* p = ws + off;
    off = (off + bytes + 255) & ~(size_t)255;
    return p;
  };
  const size_t M1 = (size_t)kB * kT;        // 16384 rows
  bf16_t* xbf     = (bf16_t*)carve(M1 * kH * 2);                    // 32 MB
  bf16_t* wqkT    = (bf16_t*)carve((size_t)kH * kH * 2);            //  2 MB
  bf16_t* wvT     = (bf16_t*)carve((size_t)kH * kH * 2);            //  2 MB
  bf16_t* woT     = (bf16_t*)carve((size_t)kH * kH * 2);            //  2 MB
  float*  qkbuf   = (float*) carve((size_t)kBh * kT * kD * 4);      // 64 MB
  bf16_t* vbuf    = (bf16_t*)carve((size_t)kBh * kT * kD * 2);      // 32 MB
  unsigned int* keybuf = (unsigned int*)carve((size_t)kBh * kNT * 4);
  int*    sticker = (int*)   carve((size_t)kBh * kNT * 4);
  float*  uo      = (float*) carve((size_t)kBh * kNHash * kT * kD * 4); // 128 MB
  float*  ulse    = (float*) carve((size_t)kBh * kNHash * kT * 4);
  bf16_t* headout = (bf16_t*)carve(M1 * kH * 2);                    // 32 MB

  // ---- 1. precision / layout prep ----
  {
    int n = (int)(M1 * kH);
    cvt_f32_to_bf16<<<(n + 255) / 256, 256, 0, stream>>>(q_in, xbf, n);
  }
  {
    int n = kH * kH;
    transpose_f32_to_bf16<<<(n + 255) / 256, 256, 0, stream>>>(wqk, wqkT, kH, kH);
    transpose_f32_to_bf16<<<(n + 255) / 256, 256, 0, stream>>>(wv,  wvT,  kH, kH);
    transpose_f32_to_bf16<<<(n + 255) / 256, 256, 0, stream>>>(wo,  woT,  kH, kH);
  }

  // ---- 2. QK and V projections (WMMA, 32x64 register blocks) ----
  {
    int waves  = (int)(M1 / 32) * (kH / 64);
    int blocks = (waves + 7) / 8;
    wmma_gemm_bf16<<<blocks, 256, 0, stream>>>(xbf, wqkT, (int)M1, kH, kH,
                                               qkbuf, nullptr, 0);
    wmma_gemm_bf16<<<blocks, 256, 0, stream>>>(xbf, wvT,  (int)M1, kH, kH,
                                               nullptr, vbuf, 1);
  }

  // ---- 3. LSH bucketing ----
  {
    int n = kBh * kT;
    lsh_buckets<<<(n + 255) / 256, 256, 0, stream>>>(qkbuf, rot, keybuf);
  }

  // ---- 4. per-head bitonic sort ----
  lsh_sort<<<kBh, 1024, 0, stream>>>(keybuf, sticker);

  // ---- 5. chunked attention (WMMA) ----
  lsh_attention<<<kBh * kC, 128, 0, stream>>>(qkbuf, vbuf, sticker, pad, uo, ulse);

  // ---- 6. combine hash rounds ----
  {
    int n = kBh * kT;
    lsh_combine<<<(n + 255) / 256, 256, 0, stream>>>(uo, ulse, headout);
  }

  // ---- 7. output projection (WMMA, 32x64 register blocks) -> d_out ----
  {
    int waves  = (int)(M1 / 32) * (kH / 64);
    int blocks = (waves + 7) / 8;
    wmma_gemm_bf16<<<blocks, 256, 0, stream>>>(headout, woT, (int)M1, kH, kH,
                                               out, nullptr, 2);
  }
}
